// VocabularyAttention_21509196218875
// MI455X (gfx1250) — compile-verified, matrix operand order checked by reading, not run on hardware
//
#include <hip/hip_runtime.h>
#include <hip/hip_bf16.h>

// ---------------- problem constants ----------------
#define B_DIM   4
#define S_DIM   2048
#define D_DIM   1024
#define V_DIM   32000
#define ROWS    (B_DIM * S_DIM)      // 8192 query rows
#define NT      64                   // vocab tile per iteration
#define QT      16                   // query rows per block (WMMA M)
#define DCHUNK  128                  // D columns owned by each wave (8*128 = 1024)

typedef __attribute__((ext_vector_type(16))) __bf16 v16bf;
typedef __attribute__((ext_vector_type(8)))  float  v8f;

// ---------------------------------------------------------------------------
// WMMA 16-bit fragment load when the K dimension is contiguous in memory:
// per the CDNA5 16x32 16-bit layout each lane's 16 elements are two
// contiguous 16-byte runs -> exactly two b128 loads per lane.
// ---------------------------------------------------------------------------
__device__ __forceinline__ v16bf make_frag_k1(const __bf16* __restrict__ base,
                                              int mnstride) {
  const int lane = threadIdx.x & 31;
  const int g8   = (lane >> 4) << 3;   // +8 K for lanes 16..31
  const int mn   = lane & 15;
  const __bf16* p = base + (size_t)mn * mnstride + g8;
  union { uint4 q[2]; v16bf v; } u;
  u.q[0] = *reinterpret_cast<const uint4*>(p);        // K = g8..g8+7
  u.q[1] = *reinterpret_cast<const uint4*>(p + 16);   // K = 16+g8..23+g8
  return u.v;
}

// ---------------- prep: fp32 -> bf16 ----------------
__global__ void to_bf16_kernel(const float* __restrict__ src,
                               unsigned short* __restrict__ dst, int n) {
  int i = blockIdx.x * blockDim.x + threadIdx.x;
  const int stride = blockDim.x * gridDim.x;
  for (; i < n; i += stride) {
    __bf16 h = (__bf16)src[i];
    dst[i] = __builtin_bit_cast(unsigned short, h);
  }
}

// ---------------- prep: fp32 [V,D] -> bf16 transposed [D,V] ----------------
__global__ void transpose_bf16_kernel(const float* __restrict__ src,
                                      unsigned short* __restrict__ dst) {
  int i = blockIdx.x * blockDim.x + threadIdx.x;
  const int stride = blockDim.x * gridDim.x;
  const int n = V_DIM * D_DIM;
  for (; i < n; i += stride) {
    const int d = i / V_DIM;
    const int v = i - d * V_DIM;
    __bf16 h = (__bf16)src[(size_t)v * D_DIM + d];
    dst[i] = __builtin_bit_cast(unsigned short, h);
  }
}

// ---------------------------------------------------------------------------
// Fused: flash-style softmax over vocabulary + MLP, per 16-row query tile.
// 256 threads = 8 wave32; wave w owns D columns [w*128, w*128+128).
// ---------------------------------------------------------------------------
__global__ __launch_bounds__(256)
void vocab_attn_mlp_kernel(const unsigned short* __restrict__ E_,   // [V, D] bf16
                           const unsigned short* __restrict__ Et_,  // [D, V] bf16
                           const unsigned short* __restrict__ X_,   // [ROWS, D] bf16
                           const unsigned short* __restrict__ W_,   // [D, 2D] bf16
                           const float* __restrict__ bias,
                           float* __restrict__ out) {
  const __bf16* __restrict__ Ebf = reinterpret_cast<const __bf16*>(E_);
  const __bf16* __restrict__ Etb = reinterpret_cast<const __bf16*>(Et_);
  const __bf16* __restrict__ Xbf = reinterpret_cast<const __bf16*>(X_);
  const __bf16* __restrict__ Wbf = reinterpret_cast<const __bf16*>(W_);

  __shared__ __align__(16) __bf16 sh_x[QT][D_DIM];   // 32 KB: x tile (h first half)
  __shared__ __align__(16) __bf16 sh_a[QT][D_DIM];   // 32 KB: attn out (h second half)
  __shared__ float  sh_scores[QT][NT];               // 4 KB: cross-wave reduction
  __shared__ __align__(16) __bf16 sh_p[QT][NT];      // 2 KB: softmax probs
  __shared__ float  sh_c[QT];                        // per-row rescale factor
  __shared__ float  sh_m[QT];                        // running max
  __shared__ float  sh_l[QT];                        // running sum
  __shared__ float  sh_inv[QT];                      // 1 / final sum
  __shared__ int    sh_flag;                         // "any max updated" flag

  const int tid  = threadIdx.x;
  const int lane = tid & 31;
  const int wid  = tid >> 5;
  const int g8   = (lane >> 4) << 3;
  const int mn   = lane & 15;
  const int q0   = blockIdx.x * QT;
  const int d0w  = wid * DCHUNK;

  // ---- stage x tile (for the MLP h) ----
  {
    const uint4* src = reinterpret_cast<const uint4*>(Xbf + (size_t)q0 * D_DIM);
    uint4* dst = reinterpret_cast<uint4*>(&sh_x[0][0]);
#pragma unroll
    for (int i = 0; i < 8; ++i) dst[tid + i * 256] = src[tid + i * 256];
  }
  if (tid < QT) { sh_m[tid] = -1e30f; sh_l[tid] = 0.0f; }

  // Preload the 4 x A-fragments for this wave's D chunk straight from global.
  v16bf ax[4];
#pragma unroll
  for (int kk = 0; kk < 4; ++kk)
    ax[kk] = make_frag_k1(Xbf + (size_t)q0 * D_DIM + d0w + kk * 32, D_DIM);

  // Per-wave attention accumulator: 16 rows x 128 cols f32 (8 WMMA C tiles).
  v8f acc[8];
#pragma unroll
  for (int t = 0; t < 8; ++t) acc[t] = (v8f)(0.0f);

  __syncthreads();

  // ================= vocabulary stream: 500 tiles of 64 =================
  for (int it = 0; it < V_DIM / NT; ++it) {
    const int v0 = it * NT;
    __syncthreads();  // prev readers of sh_scores / sh_p / sh_flag done

    // zero the cross-wave score reduction buffer + flag
#pragma unroll
    for (int i = 0; i < (QT * NT) / 256; ++i)
      (&sh_scores[0][0])[tid + i * 256] = 0.0f;
    if (tid == 0) sh_flag = 0;

    if (v0 + NT < V_DIM)  // keep next E tile hot (global_prefetch_b8)
      __builtin_prefetch(Ebf + (size_t)(v0 + NT) * D_DIM + tid * 256, 0, 0);

    // ---- scores: S[16, 64] partial over this wave's D chunk ----
    // Load-all / WMMA-all in groups of 8 fragments so the loads form one
    // clause and the WMMAs drain it with staggered s_wait_loadcnt.
    v8f s[4];
#pragma unroll
    for (int ns = 0; ns < 4; ++ns) s[ns] = (v8f)(0.0f);
#pragma unroll
    for (int gi = 0; gi < 2; ++gi) {
      v16bf bf[8];
#pragma unroll
      for (int u = 0; u < 8; ++u) {            // u = ns_local*4 + kk
        const int ns = gi * 2 + (u >> 2);
        const int kk = u & 3;
        bf[u] = make_frag_k1(Ebf + (size_t)(v0 + ns * 16) * D_DIM + d0w + kk * 32,
                             D_DIM);
      }
#pragma unroll
      for (int u = 0; u < 8; ++u) {
        const int ns = gi * 2 + (u >> 2);
        s[ns] = __builtin_amdgcn_wmma_f32_16x16x32_bf16(false, ax[u & 3], false,
                                                        bf[u], (short)0, s[ns],
                                                        false, false);
      }
    }
    __syncthreads();  // zeroed buffer visible everywhere

    // cross-wave reduction of D-chunk partials via LDS float atomics
#pragma unroll
    for (int ns = 0; ns < 4; ++ns)
#pragma unroll
      for (int j = 0; j < 8; ++j)
        atomicAdd(&sh_scores[j + g8][ns * 16 + mn], s[ns][j]);
    __syncthreads();

    // ---- online softmax, all 8 waves: wave w owns rows 2w and 2w+1 ----
    {
      const int half = lane >> 4;            // 0/1 -> which of the two rows
      const int r    = 2 * wid + half;
      const int c0   = lane & 15;            // 16 lanes x 4 columns per row
      float v0f = sh_scores[r][c0];
      float v1f = sh_scores[r][c0 + 16];
      float v2f = sh_scores[r][c0 + 32];
      float v3f = sh_scores[r][c0 + 48];
      float mx = fmaxf(fmaxf(v0f, v1f), fmaxf(v2f, v3f));
#pragma unroll
      for (int msk = 8; msk >= 1; msk >>= 1)   // stays within 16-lane half
        mx = fmaxf(mx, __shfl_xor(mx, msk, 32));
      const float mold = sh_m[r];
      const float mnew = fmaxf(mold, mx);
      const float p0 = __expf(v0f - mnew);
      const float p1 = __expf(v1f - mnew);
      const float p2 = __expf(v2f - mnew);
      const float p3 = __expf(v3f - mnew);
      float sum = (p0 + p1) + (p2 + p3);
#pragma unroll
      for (int msk = 8; msk >= 1; msk >>= 1)
        sum += __shfl_xor(sum, msk, 32);
      sh_p[r][c0]      = (__bf16)p0;
      sh_p[r][c0 + 16] = (__bf16)p1;
      sh_p[r][c0 + 32] = (__bf16)p2;
      sh_p[r][c0 + 48] = (__bf16)p3;
      if (c0 == 0) {
        const float corr = __expf(mold - mnew);
        sh_c[r] = corr;
        sh_l[r] = sh_l[r] * corr + sum;
        sh_m[r] = mnew;
        if (mnew > mold) atomicOr(&sh_flag, 1);
      }
    }
    __syncthreads();

    // ---- rescale accumulator only when some running max moved ----
    if (sh_flag) {
#pragma unroll
      for (int j = 0; j < 8; ++j) {
        const float cf = sh_c[j + g8];
#pragma unroll
        for (int t = 0; t < 8; ++t) acc[t][j] *= cf;
      }
    }

    // ---- A += P[16,64] @ E[64, d-chunk] ----
    const v16bf pf0 = make_frag_k1(&sh_p[0][0], NT);       // K = v0..v0+31
    const v16bf pf1 = make_frag_k1(&sh_p[0][32], NT);      // K = v0+32..v0+63
#pragma unroll
    for (int th = 0; th < 2; ++th) {           // 4 output tiles per group
      v16bf bf[8];
#pragma unroll
      for (int u = 0; u < 8; ++u) {            // u = t_local*2 + khalf
        const int t  = th * 4 + (u >> 1);
        const int kv = (u & 1) * 32;
        bf[u] = make_frag_k1(Etb + (size_t)(d0w + t * 16) * V_DIM + v0 + kv,
                             V_DIM);
      }
#pragma unroll
      for (int u = 0; u < 8; ++u) {
        const int t = th * 4 + (u >> 1);
        acc[t] = __builtin_amdgcn_wmma_f32_16x16x32_bf16(false, (u & 1) ? pf1 : pf0,
                                                         false, bf[u], (short)0,
                                                         acc[t], false, false);
      }
    }
  }

  __syncthreads();
  if (tid < QT) sh_inv[tid] = 1.0f / sh_l[tid];
  __syncthreads();

  // normalized attention output -> bf16 -> LDS (second half of h)
#pragma unroll
  for (int j = 0; j < 8; ++j) {
    const int row = j + g8;
    const float inv = sh_inv[row];
#pragma unroll
    for (int t = 0; t < 8; ++t)
      sh_a[row][d0w + t * 16 + mn] = (__bf16)(acc[t][j] * inv);
  }
  __syncthreads();

  // ================= MLP: out[16,1024] = [x|A][16,2048] @ W^T =================
  v8f acc2[8];
#pragma unroll
  for (int t = 0; t < 8; ++t) acc2[t] = (v8f)(0.0f);

  for (int kk = 0; kk < (2 * D_DIM) / 32; ++kk) {  // 64 K-steps over e
    const int e0 = kk * 32;
    const __bf16* hbase = (e0 < D_DIM) ? &sh_x[0][e0] : &sh_a[0][e0 - D_DIM];
    const v16bf a = make_frag_k1(hbase, D_DIM);     // 2x ds_load_b128
    v16bf bf[8];
#pragma unroll
    for (int t = 0; t < 8; ++t)
      bf[t] = make_frag_k1(Wbf + (size_t)(d0w + t * 16) * (2 * D_DIM) + e0,
                           2 * D_DIM);
#pragma unroll
    for (int t = 0; t < 8; ++t)
      acc2[t] = __builtin_amdgcn_wmma_f32_16x16x32_bf16(false, a, false, bf[t],
                                                        (short)0, acc2[t],
                                                        false, false);
  }

  // epilogue: bias + f32 store
#pragma unroll
  for (int j = 0; j < 8; ++j) {
    const int row = j + g8;
#pragma unroll
    for (int t = 0; t < 8; ++t) {
      const int col = d0w + t * 16 + mn;
      out[(size_t)(q0 + row) * D_DIM + col] = acc2[t][j] + bias[col];
    }
  }
}

// ---------------------------------------------------------------------------
extern "C" void kernel_launch(void* const* d_in, const int* in_sizes, int n_in,
                              void* d_out, int out_size, void* d_ws, size_t ws_size,
                              hipStream_t stream) {
  (void)in_sizes; (void)n_in; (void)out_size; (void)ws_size;
  const float* x   = (const float*)d_in[0];  // [8192, 1024]
  const float* emb = (const float*)d_in[1];  // [32000, 1024]
  const float* w   = (const float*)d_in[2];  // [1024, 2048]
  const float* b   = (const float*)d_in[3];  // [1024]

  char* ws = (char*)d_ws;
  const size_t EV = (size_t)V_DIM * D_DIM * 2;     // 65,536,000 B
  unsigned short* Ebf = (unsigned short*)ws;                       // [V, D]
  unsigned short* Etb = (unsigned short*)(ws + EV);                // [D, V]
  unsigned short* Xbf = (unsigned short*)(ws + 2 * EV);            // [ROWS, D]
  unsigned short* Wbf = (unsigned short*)(ws + 2 * EV
                                             + (size_t)ROWS * D_DIM * 2);

  hipLaunchKernelGGL(to_bf16_kernel, dim3(2048), dim3(256), 0, stream,
                     emb, Ebf, V_DIM * D_DIM);
  hipLaunchKernelGGL(transpose_bf16_kernel, dim3(2048), dim3(256), 0, stream,
                     emb, Etb);
  hipLaunchKernelGGL(to_bf16_kernel, dim3(1024), dim3(256), 0, stream,
                     x, Xbf, ROWS * D_DIM);
  hipLaunchKernelGGL(to_bf16_kernel, dim3(512), dim3(256), 0, stream,
                     w, Wbf, D_DIM * 2 * D_DIM);

  hipLaunchKernelGGL(vocab_attn_mlp_kernel, dim3(ROWS / QT), dim3(256), 0, stream,
                     Ebf, Etb, Xbf, Wbf, b, (float*)d_out);
}
